// TriplaneField_67585605370029
// MI455X (gfx1250) — matrix operand order repaired and checked by reading, not compiled, not run on hardware
//
#include <hip/hip_runtime.h>
#include <hip/hip_bf16.h>

// ---------------------------------------------------------------------------
// TriplaneField fused forward for gfx1250 (MI455X, wave32, WMMA)
//   P points: posenc + triplane gather -> 62-dim -> MLP 256/256/256
//   Whole MLP fused per 32-point tile in LDS; GEMMs on v_wmma_f32_16x16x32_f16
// ---------------------------------------------------------------------------

typedef __attribute__((ext_vector_type(16))) _Float16 v16h;
typedef __attribute__((ext_vector_type(8)))  float    v8f;

#define RES     512
#define CDIM    32
#define HID     256
#define KPAD1   64          // D_IN=62 padded to 64
#define BM      32          // points per block
#define H0_STR  72          // padded LDS row stride (halves) for h0
#define H_STR   264         // padded LDS row stride (halves) for h1/h2

__device__ __forceinline__ unsigned short f2h(float v) {
    union { _Float16 h; unsigned short u; } cv;
    cv.h = (_Float16)v;
    return cv.u;
}

__device__ __forceinline__ v8f wmma16(v16h a, v16h b, v8f c) {
    // (neg_a, A, neg_b, B, c_mod, C, reuse_a, reuse_b)
    return __builtin_amdgcn_wmma_f32_16x16x32_f16(false, a, false, b, (short)0, c,
                                                  false, false);
}

// A-fragment (16x32 f16) from LDS row-major [m][k] with row stride `stride`
// lane layout: m = lane&15, half = lane>>4; VGPR v holds K = 16*(v>>2)+8*half+2*(v&3)+{0,1}
__device__ __forceinline__ v16h load_a_frag(const unsigned short* hbuf, int stride,
                                            int m_base, int kbase, int lane) {
    const int half = lane >> 4;
    const int m    = m_base + (lane & 15);
    const unsigned short* row = hbuf + m * stride;
    union { v16h h; unsigned u[8]; } f;
#pragma unroll
    for (int v = 0; v < 8; ++v) {
        int k = kbase + ((v >> 2) << 4) + (half << 3) + ((v & 3) << 1);
        f.u[v] = *(const unsigned*)(row + k);
    }
    return f.h;
}

// B-fragment (32x16 f16) from global Wt[n][k] (K contiguous), row stride kstride
// lane layout: n = lane&15, half = lane>>4; VGPR v holds K = 16*(v>>2)+8*half+2*(v&3)+{0,1}
__device__ __forceinline__ v16h load_b_frag(const unsigned short* __restrict__ wt,
                                            int kstride, int n0, int kbase, int lane) {
    const int half = lane >> 4;
    const int n    = n0 + (lane & 15);
    const unsigned short* col = wt + n * kstride;
    union { v16h h; unsigned u[8]; } f;
#pragma unroll
    for (int v = 0; v < 8; ++v) {
        int k = kbase + ((v >> 2) << 4) + (half << 3) + ((v & 3) << 1);
        f.u[v] = *(const unsigned*)(col + k);
    }
    return f.h;
}

// Bilinear taps for grid_sample (border padding, align_corners=True), 512x512
__device__ __forceinline__ void bil(float cx, float cy, int* o, float* w) {
    float ix = (cx + 1.0f) * 0.5f * (float)(RES - 1);
    float iy = (cy + 1.0f) * 0.5f * (float)(RES - 1);
    float fx = floorf(ix), fy = floorf(iy);
    float wx = ix - fx,    wy = iy - fy;
    int x0 = (int)fminf(fmaxf(fx,        0.0f), (float)(RES - 1));
    int x1 = (int)fminf(fmaxf(fx + 1.0f, 0.0f), (float)(RES - 1));
    int y0 = (int)fminf(fmaxf(fy,        0.0f), (float)(RES - 1));
    int y1 = (int)fminf(fmaxf(fy + 1.0f, 0.0f), (float)(RES - 1));
    o[0] = y0 * RES + x0;  o[1] = y0 * RES + x1;
    o[2] = y1 * RES + x0;  o[3] = y1 * RES + x1;
    w[0] = (1.0f - wx) * (1.0f - wy);
    w[1] = wx * (1.0f - wy);
    w[2] = (1.0f - wx) * wy;
    w[3] = wx * wy;
}

// ---------------------------------------------------------------------------
// Weight prep: f32 [k][n] -> f16 [n][k] (K contiguous), W1 zero-padded 62->64
// ---------------------------------------------------------------------------
__global__ void __launch_bounds__(256)
triplane_prep_weights(const float* __restrict__ W1, const float* __restrict__ W2,
                      const float* __restrict__ W3,
                      unsigned short* __restrict__ wt1,
                      unsigned short* __restrict__ wt2,
                      unsigned short* __restrict__ wt3) {
    const int n = blockIdx.x;      // 0..255 output column
    const int t = threadIdx.x;     // 0..255 k index
    if (t < KPAD1) {
        float v = (t < 62) ? W1[t * HID + n] : 0.0f;
        wt1[n * KPAD1 + t] = f2h(v);
    }
    wt2[n * HID + t] = f2h(W2[t * HID + n]);
    wt3[n * HID + t] = f2h(W3[t * HID + n]);
}

// ---------------------------------------------------------------------------
// Fused forward: one 32-point tile per block, 256 threads = 8 waves
// ---------------------------------------------------------------------------
__global__ void __launch_bounds__(256)
triplane_mlp_fused(const float* __restrict__ x,
                   const float* __restrict__ mean,
                   const float* __restrict__ scale,
                   const float* __restrict__ planes,
                   const float* __restrict__ b1,
                   const float* __restrict__ b2,
                   const float* __restrict__ b3,
                   const unsigned short* __restrict__ wt1,
                   const unsigned short* __restrict__ wt2,
                   const unsigned short* __restrict__ wt3,
                   float* __restrict__ out, int Ptot) {
    __shared__ float          ptsS[BM * 3];
    __shared__ unsigned short h0s[BM * H0_STR];   //  4.6 KB
    __shared__ unsigned short h1s[BM * H_STR];    // 16.9 KB
    __shared__ unsigned short h2s[BM * H_STR];    // 16.9 KB

    const int tid  = threadIdx.x;
    const int lane = tid & 31;
    const int wave = tid >> 5;
    const int P0   = blockIdx.x * BM;

    // ---------------- Phase A1: normalize + positional encoding ------------
    if (tid < BM) {
        int pidx = P0 + tid;
        if (pidx >= Ptot) pidx = Ptot - 1;        // clamp (no harm, store guarded)
        float xv[3];
#pragma unroll
        for (int c = 0; c < 3; ++c) {
            float v = (x[pidx * 3 + c] - mean[c]) / scale[c];
            xv[c] = v;
            ptsS[tid * 3 + c] = v;
        }
        unsigned short* row = h0s + tid * H0_STR;
#pragma unroll
        for (int c = 0; c < 3; ++c) {
#pragma unroll
            for (int f = 0; f < 5; ++f) {
                float ph = xv[c] * (float)(1 << f);
                int k = 2 * (c * 5 + f);
                row[k]     = f2h(sinf(ph));
                row[k + 1] = f2h(cosf(ph));
            }
        }
        row[62] = 0;  row[63] = 0;                // K padding
    }
    __syncthreads();

    // ---------------- Phase A2: triplane bilinear gather (all 256 thr) -----
    {
        const int p = tid >> 3;                   // point in tile
        const int j = tid & 7;                    // channel group
        const float px = ptsS[p * 3 + 0];
        const float py = ptsS[p * 3 + 1];
        const float pz = ptsS[p * 3 + 2];
        int   o[3][4];
        float w[3][4];
        bil(px, py, o[0], w[0]);                  // plane xy
        bil(px, pz, o[1], w[1]);                  // plane xz
        bil(py, pz, o[2], w[2]);                  // plane yz
        unsigned short* row = h0s + p * H0_STR + 30;
#pragma unroll
        for (int q = 0; q < 4; ++q) {
            const int ch = j + q * 8;
            float acc = 0.0f;
#pragma unroll
            for (int pl = 0; pl < 3; ++pl) {
                const float* base = planes + ((pl * CDIM + ch) * (RES * RES));
                acc += base[o[pl][0]] * w[pl][0] + base[o[pl][1]] * w[pl][1]
                     + base[o[pl][2]] * w[pl][2] + base[o[pl][3]] * w[pl][3];
            }
            row[ch] = f2h(acc * (1.0f / 3.0f));
        }
    }
    __syncthreads();

    // Wave tiling: 2 m-strips x 4 n-blocks (each n-block = 4 tiles of 16 cols)
    const int m_base = (wave & 1) << 4;           // 0 or 16
    const int n_base = (wave >> 1) << 6;          // 0,64,128,192
    const int half   = lane >> 4;
    const int ln     = lane & 15;

    v8f acc[4];

    // ---------------- GEMM1: h0(32x64) x W1(64x256) -> h1 ------------------
#pragma unroll
    for (int t = 0; t < 4; ++t) acc[t] = (v8f){0,0,0,0,0,0,0,0};
#pragma unroll
    for (int kb = 0; kb < KPAD1; kb += 32) {
        v16h a = load_a_frag(h0s, H0_STR, m_base, kb, lane);
#pragma unroll
        for (int t = 0; t < 4; ++t) {
            v16h b = load_b_frag(wt1, KPAD1, n_base + t * 16, kb, lane);
            acc[t] = wmma16(a, b, acc[t]);
        }
    }
#pragma unroll
    for (int t = 0; t < 4; ++t) {
        const int ncol = n_base + t * 16 + ln;
        const float bias = b1[ncol];
#pragma unroll
        for (int r = 0; r < 8; ++r) {
            const int m = m_base + r + (half << 3);
            float v = acc[t][r] + bias;
            v = (v > 0.0f) ? v : 0.01f * v;       // leaky relu
            h1s[m * H_STR + ncol] = f2h(v);
        }
    }
    __syncthreads();

    // ---------------- GEMM2: h1(32x256) x W2(256x256) -> h2 ----------------
#pragma unroll
    for (int t = 0; t < 4; ++t) acc[t] = (v8f){0,0,0,0,0,0,0,0};
#pragma unroll
    for (int kb = 0; kb < HID; kb += 32) {
        v16h a = load_a_frag(h1s, H_STR, m_base, kb, lane);
#pragma unroll
        for (int t = 0; t < 4; ++t) {
            v16h b = load_b_frag(wt2, HID, n_base + t * 16, kb, lane);
            acc[t] = wmma16(a, b, acc[t]);
        }
    }
#pragma unroll
    for (int t = 0; t < 4; ++t) {
        const int ncol = n_base + t * 16 + ln;
        const float bias = b2[ncol];
#pragma unroll
        for (int r = 0; r < 8; ++r) {
            const int m = m_base + r + (half << 3);
            float v = acc[t][r] + bias;
            v = (v > 0.0f) ? v : 0.01f * v;
            h2s[m * H_STR + ncol] = f2h(v);
        }
    }
    __syncthreads();

    // ---------------- GEMM3: h2(32x256) x W3(256x256) -> out ---------------
#pragma unroll
    for (int t = 0; t < 4; ++t) acc[t] = (v8f){0,0,0,0,0,0,0,0};
#pragma unroll
    for (int kb = 0; kb < HID; kb += 32) {
        v16h a = load_a_frag(h2s, H_STR, m_base, kb, lane);
#pragma unroll
        for (int t = 0; t < 4; ++t) {
            v16h b = load_b_frag(wt3, HID, n_base + t * 16, kb, lane);
            acc[t] = wmma16(a, b, acc[t]);
        }
    }
#pragma unroll
    for (int t = 0; t < 4; ++t) {
        const int ncol = n_base + t * 16 + ln;
        const float bias = b3[ncol];
#pragma unroll
        for (int r = 0; r < 8; ++r) {
            const int m = m_base + r + (half << 3);
            const int prow = P0 + m;
            if (prow < Ptot) out[(size_t)prow * HID + ncol] = acc[t][r] + bias;
        }
    }
}

// ---------------------------------------------------------------------------
extern "C" void kernel_launch(void* const* d_in, const int* in_sizes, int n_in,
                              void* d_out, int out_size, void* d_ws, size_t ws_size,
                              hipStream_t stream) {
    const float* x      = (const float*)d_in[0];
    const float* mean   = (const float*)d_in[1];
    const float* scale  = (const float*)d_in[2];
    // d_in[3..5] = pca_rot/pca_t/pca_scale (unused on this path)
    const float* planes = (const float*)d_in[6];
    const float* W1     = (const float*)d_in[7];
    const float* b1     = (const float*)d_in[8];
    const float* W2     = (const float*)d_in[9];
    const float* b2     = (const float*)d_in[10];
    const float* W3     = (const float*)d_in[11];
    const float* b3     = (const float*)d_in[12];
    float* out          = (float*)d_out;

    unsigned short* ws  = (unsigned short*)d_ws;
    unsigned short* wt1 = ws;                         // 256*64  halves
    unsigned short* wt2 = ws + 256 * KPAD1;           // 256*256 halves
    unsigned short* wt3 = wt2 + HID * HID;            // 256*256 halves

    triplane_prep_weights<<<HID, 256, 0, stream>>>(W1, W2, W3, wt1, wt2, wt3);

    const int Ptot   = in_sizes[0] / 3;               // 262144
    const int blocks = (Ptot + BM - 1) / BM;          // 8192
    triplane_mlp_fused<<<blocks, 256, 0, stream>>>(x, mean, scale, planes,
                                                   b1, b2, b3, wt1, wt2, wt3,
                                                   out, Ptot);
}
